// GNNBackbone_35347580846470
// MI455X (gfx1250) — compile-verified, matrix operand order checked by reading, not run on hardware
//
#include <hip/hip_runtime.h>
#include <hip/hip_bf16.h>
#include <math.h>

#define N_NODES  10000
#define N_EDGES  320000
#define N_GRAPHS 64
#define HID      256
#define MPAD     10112   // 79 * 128, padded row count for guard-free WMMA GEMM

typedef __attribute__((ext_vector_type(16))) __bf16 v16bf;
typedef __attribute__((ext_vector_type(8)))  float  v8f;

union Frag16 { v16bf v; uint4 u[2]; };

__device__ __forceinline__ float silu_f(float v) { return v / (1.0f + __expf(-v)); }

// ---------------- prep: Wt[n][k] = bf16(W[k][n]) ----------------
__global__ void k_transpose_w(const float* __restrict__ W, __hip_bfloat16* __restrict__ Wt) {
    int i = blockIdx.x * blockDim.x + threadIdx.x;   // 256*256 threads
    int n = i >> 8, k = i & 255;
    Wt[n * HID + k] = __float2bfloat16(W[k * HID + n]);
}

// ---------------- embed: h = silu(x * Wemb + bemb); zero-pad rows ----------------
__global__ void k_embed(const float* __restrict__ x, const float* __restrict__ Wemb,
                        const float* __restrict__ bemb,
                        float* __restrict__ h, __hip_bfloat16* __restrict__ hb) {
    int i = blockIdx.x * blockDim.x + threadIdx.x;   // MPAD*HID threads
    int n = i >> 8, j = i & 255;
    float v = 0.0f;
    if (n < N_NODES) v = silu_f(x[n] * Wemb[j] + bemb[j]);
    h[i]  = v;
    hb[i] = __float2bfloat16(v);
}

// ---------------- WMMA GEMM: C = act(A[MPAD x 256] @ W + bias) ----------------
// A bf16 row-major, Wt = W^T bf16 row-major (so B fragments load like A fragments).
// Block = 256 threads = 8 waves -> 128(M) x 128(N) tile; wave = 32(M) x 64(N).
template<bool RELU_BF16_OUT>
__global__ __launch_bounds__(256)
void k_gemm(const __hip_bfloat16* __restrict__ A,
            const __hip_bfloat16* __restrict__ Wt,
            const float* __restrict__ bias,
            float* __restrict__ Cf,            // used when !RELU_BF16_OUT
            __hip_bfloat16* __restrict__ Cb) { // used when  RELU_BF16_OUT
    const int lane  = threadIdx.x & 31;
    const int wave  = threadIdx.x >> 5;
    const int half  = lane >> 4;       // 0: K-low 8, 1: K-high 8 (per ISA 16-bit A layout)
    const int l16   = lane & 15;
    const int mWave = blockIdx.x * 128 + (wave >> 1) * 32;
    const int nWave = blockIdx.y * 128 + (wave & 1) * 64;

    v8f acc[2][4];
    v8f zero = {};
#pragma unroll
    for (int mi = 0; mi < 2; ++mi)
#pragma unroll
        for (int ni = 0; ni < 4; ++ni) acc[mi][ni] = zero;

    for (int kb = 0; kb < HID; kb += 32) {
        Frag16 afr[2], bfr[4];
#pragma unroll
        for (int mi = 0; mi < 2; ++mi) {
            const __hip_bfloat16* p = A + (size_t)(mWave + mi * 16 + l16) * HID + kb + half * 8;
            afr[mi].u[0] = *(const uint4*)(p);
            afr[mi].u[1] = *(const uint4*)(p + 16);
            if (kb + 32 < HID) __builtin_prefetch(p + 32, 0, 3);  // global_prefetch_b8
        }
#pragma unroll
        for (int ni = 0; ni < 4; ++ni) {
            const __hip_bfloat16* p = Wt + (size_t)(nWave + ni * 16 + l16) * HID + kb + half * 8;
            bfr[ni].u[0] = *(const uint4*)(p);
            bfr[ni].u[1] = *(const uint4*)(p + 16);
        }
#pragma unroll
        for (int mi = 0; mi < 2; ++mi)
#pragma unroll
            for (int ni = 0; ni < 4; ++ni)
                acc[mi][ni] = __builtin_amdgcn_wmma_f32_16x16x32_bf16(
                    false, afr[mi].v, false, bfr[ni].v,
                    (short)0, acc[mi][ni], false, false);
    }

    // C/D layout: vgpr v, lane: row = half*8 + v, col = lane&15 (within 16x16 tile)
#pragma unroll
    for (int mi = 0; mi < 2; ++mi) {
#pragma unroll
        for (int ni = 0; ni < 4; ++ni) {
            const int col = nWave + ni * 16 + l16;
            const float bv = bias[col];
#pragma unroll
            for (int v = 0; v < 8; ++v) {
                const int row = mWave + mi * 16 + half * 8 + v;
                const float val = acc[mi][ni][v] + bv;
                if (RELU_BF16_OUT)
                    Cb[(size_t)row * HID + col] = __float2bfloat16(fmaxf(val, 0.0f));
                else
                    Cf[(size_t)row * HID + col] = val;
            }
        }
    }
}

// ---------------- scatter-max over edges ----------------
__global__ void k_fill_neginf(float* __restrict__ agg) {
    int i = blockIdx.x * blockDim.x + threadIdx.x;   // N_NODES*HID
    if (i < N_NODES * HID) agg[i] = -INFINITY;
}

__global__ void k_scatter_max(const int* __restrict__ ei, const float* __restrict__ m,
                              float* __restrict__ agg) {
    long long t = (long long)blockIdx.x * blockDim.x + threadIdx.x;  // N_EDGES*64
    int e  = (int)(t >> 6);
    if (e >= N_EDGES) return;
    int j4 = ((int)t & 63) * 4;
    int src = ei[e];             // edge_index row 0
    int dst = ei[N_EDGES + e];   // edge_index row 1
    const float4 v = *(const float4*)(m + (size_t)src * HID + j4);
    float* a = agg + (size_t)dst * HID + j4;
    atomicMax(a + 0, v.x);
    atomicMax(a + 1, v.y);
    atomicMax(a + 2, v.z);
    atomicMax(a + 3, v.w);
}

// ---------------- residual combine: h += silu(where(agg==-inf, 0, agg)) ----------------
__global__ void k_combine(const float* __restrict__ agg, float* __restrict__ h,
                          __hip_bfloat16* __restrict__ hb) {
    int i = blockIdx.x * blockDim.x + threadIdx.x;   // N_NODES*HID
    if (i >= N_NODES * HID) return;
    float a = agg[i];
    if (a == -INFINITY) a = 0.0f;
    float v = h[i] + silu_f(a);
    h[i]  = v;
    hb[i] = __float2bfloat16(v);
}

// ---------------- pooling ----------------
__global__ void k_zero_f32(float* __restrict__ p, int n) {
    int i = blockIdx.x * blockDim.x + threadIdx.x;
    if (i < n) p[i] = 0.0f;
}

__global__ void k_pool_acc(const float* __restrict__ h, const int* __restrict__ batch,
                           float* __restrict__ gsum, float* __restrict__ gcnt) {
    int i = blockIdx.x * blockDim.x + threadIdx.x;   // N_NODES*HID
    if (i >= N_NODES * HID) return;
    int n = i >> 8, j = i & 255;
    int g = batch[n];
    atomicAdd(&gsum[g * HID + j], h[i]);
    if (j == 0) atomicAdd(&gcnt[g], 1.0f);
}

__global__ void k_pool_final(const float* __restrict__ gsum, const float* __restrict__ gcnt,
                             float* __restrict__ out) {
    int i = blockIdx.x * blockDim.x + threadIdx.x;   // N_GRAPHS*HID
    if (i >= N_GRAPHS * HID) return;
    int g = i >> 8;
    out[i] = gsum[i] / fmaxf(gcnt[g], 1.0f);
}

extern "C" void kernel_launch(void* const* d_in, const int* in_sizes, int n_in,
                              void* d_out, int out_size, void* d_ws, size_t ws_size,
                              hipStream_t stream) {
    const float* x     = (const float*)d_in[0];
    const int*   ei    = (const int*)d_in[1];    // [2, N_EDGES] int32
    const int*   batch = (const int*)d_in[2];
    const float* Wemb  = (const float*)d_in[3];
    const float* bemb  = (const float*)d_in[4];
    const float* W1a   = (const float*)d_in[5];
    const float* b1a   = (const float*)d_in[6];
    const float* W1b   = (const float*)d_in[7];
    const float* b1b   = (const float*)d_in[8];
    const float* W2a   = (const float*)d_in[9];
    const float* b2a   = (const float*)d_in[10];
    const float* W2b   = (const float*)d_in[11];
    const float* b2b   = (const float*)d_in[12];
    float* out = (float*)d_out;

    // carve workspace (~42 MB total)
    char* ws = (char*)d_ws;
    auto carve = [&](size_t bytes) -> void* {
        void* p = (void*)ws;
        ws += (bytes + 255) & ~(size_t)255;
        return p;
    };
    float*          h    = (float*)carve((size_t)MPAD * HID * 4);
    __hip_bfloat16* hb   = (__hip_bfloat16*)carve((size_t)MPAD * HID * 2);
    __hip_bfloat16* tb   = (__hip_bfloat16*)carve((size_t)MPAD * HID * 2);
    float*          msg  = (float*)carve((size_t)MPAD * HID * 4);
    float*          agg  = (float*)carve((size_t)N_NODES * HID * 4);
    __hip_bfloat16* Wt1a = (__hip_bfloat16*)carve((size_t)HID * HID * 2);
    __hip_bfloat16* Wt1b = (__hip_bfloat16*)carve((size_t)HID * HID * 2);
    __hip_bfloat16* Wt2a = (__hip_bfloat16*)carve((size_t)HID * HID * 2);
    __hip_bfloat16* Wt2b = (__hip_bfloat16*)carve((size_t)HID * HID * 2);
    float*          gsum = (float*)carve((size_t)N_GRAPHS * HID * 4);
    float*          gcnt = (float*)carve((size_t)N_GRAPHS * 4);

    const int B = 256;
    const int gridNH   = (N_NODES * HID + B - 1) / B;       // 10000
    const int gridMH   = (MPAD * HID) / B;                  // 10112
    const int gridW    = (HID * HID) / B;                   // 256
    const int gridEdge = (int)(((long long)N_EDGES * 64 + B - 1) / B); // 80000
    dim3 gemmGrid(MPAD / 128, HID / 128);                   // (79, 2)

    // weight prep
    k_transpose_w<<<gridW, B, 0, stream>>>(W1a, Wt1a);
    k_transpose_w<<<gridW, B, 0, stream>>>(W1b, Wt1b);
    k_transpose_w<<<gridW, B, 0, stream>>>(W2a, Wt2a);
    k_transpose_w<<<gridW, B, 0, stream>>>(W2b, Wt2b);

    // embed
    k_embed<<<gridMH, B, 0, stream>>>(x, Wemb, bemb, h, hb);

    // two conv layers (MLP hoisted to nodes, scatter-max over edges)
    const __hip_bfloat16* Wa[2] = {Wt1a, Wt2a};
    const __hip_bfloat16* Wb[2] = {Wt1b, Wt2b};
    const float* ba[2] = {b1a, b2a};
    const float* bb[2] = {b1b, b2b};
    for (int layer = 0; layer < 2; ++layer) {
        k_gemm<true ><<<gemmGrid, B, 0, stream>>>(hb, Wa[layer], ba[layer], nullptr, tb);
        k_gemm<false><<<gemmGrid, B, 0, stream>>>(tb, Wb[layer], bb[layer], msg, nullptr);
        k_fill_neginf<<<gridNH, B, 0, stream>>>(agg);
        k_scatter_max<<<gridEdge, B, 0, stream>>>(ei, msg, agg);
        k_combine<<<gridNH, B, 0, stream>>>(agg, h, hb);
    }

    // global mean pool
    k_zero_f32<<<(N_GRAPHS * HID + B - 1) / B, B, 0, stream>>>(gsum, N_GRAPHS * HID);
    k_zero_f32<<<1, B, 0, stream>>>(gcnt, N_GRAPHS);
    k_pool_acc<<<gridNH, B, 0, stream>>>(h, batch, gsum, gcnt);
    k_pool_final<<<(N_GRAPHS * HID + B - 1) / B, B, 0, stream>>>(gsum, gcnt, out);
}
